// Block_53841710022747
// MI455X (gfx1250) — matrix operand order repaired and checked by reading, not compiled, not run on hardware
//
#include <hip/hip_runtime.h>
#include <math.h>

// ---------------- problem constants ----------------
#define B_    2
#define T_    2048
#define E_    1024
#define NH_   16
#define HD_   64
#define NEXP_ 8
#define CAP_  1024            // ceil(2.0 * 4096 / 8)
#define NTOK_ (B_*T_)         // 4096
#define FF_   (4*E_)          // 4096

typedef __attribute__((ext_vector_type(16))) __bf16 v16bf;
typedef __attribute__((ext_vector_type(8)))  float  v8f;

// ---------------- WMMA helpers (CDNA5, wave32) ----------------
__device__ inline v8f wmma_bf16(v16bf a, v16bf b, v8f c) {
  // D = A(16x32 bf16) * B(32x16 bf16) + C(16x16 f32)
  return __builtin_amdgcn_wmma_f32_16x16x32_bf16(false, a, false, b, (short)0, c,
                                                 false, false);
}

__device__ inline v8f zero8() {
  v8f z;
#pragma unroll
  for (int i = 0; i < 8; ++i) z[i] = 0.f;
  return z;
}

// A fragment (16x32 bf16): lane&15 = row m; lanes<16 hold K {0..15},
// lanes>=16 hold K {16..31}. 'row' points at A[m][0], contiguous K.
// Per lane this is two contiguous 16B chunks -> two global_load_b128.
__device__ inline v16bf afrag_bf16(const __bf16* __restrict__ row, int kbase, int lane) {
  v16bf a;
  const int kg = (lane >> 4) & 1;
#pragma unroll
  for (int j = 0; j < 8; ++j) {
    const int k0 = kbase + kg * 16 + 2 * j;
    a[2 * j]     = row[k0];
    a[2 * j + 1] = row[k0 + 1];
  }
  return a;
}

// B fragment (32x16), B[k][n] = Wn[k], Wn = row n of an N-major (N x K) bf16 weight.
// Lane layout: n = lane&15; lanes<16 carry K kbase..kbase+15, lanes>=16 kbase+16..+31.
// Per lane: 16 contiguous bf16 = 32B -> two global_load_b128.
__device__ inline v16bf bfrag_rowT_bf16(const __bf16* __restrict__ Wn, int kbase, int lane) {
  v16bf b;
  const int kb = kbase + ((lane < 16) ? 0 : 16);
#pragma unroll
  for (int j = 0; j < 16; ++j) b[j] = Wn[kb + j];
  return b;
}
// B fragment, B[k][n] = W[k*ldb + n] (row-major K x N, bf16). Used for V in attention.
__device__ inline v16bf bfrag_col_bf16(const __bf16* __restrict__ W, int ldb, int n,
                                       int kbase, int lane) {
  v16bf b;
  const int kb = kbase + ((lane < 16) ? 0 : 16);
#pragma unroll
  for (int j = 0; j < 16; ++j) b[j] = W[(size_t)(kb + j) * ldb + n];
  return b;
}

// ---------------- one-time weight conversion: f32 -> bf16 (same layout) ---------------
__global__ void cvt_bf16_kernel(const float* __restrict__ src, __bf16* __restrict__ dst,
                                int n4) {  // n4 = elements/4
  const int i = blockIdx.x * 256 + threadIdx.x;
  if (i < n4) {
    const float4 v = ((const float4*)src)[i];
    __bf16* d = dst + (size_t)i * 4;
    d[0] = (__bf16)v.x; d[1] = (__bf16)v.y; d[2] = (__bf16)v.z; d[3] = (__bf16)v.w;
  }
}

// ------------- one-time transpose+convert: src (R x C) f32 -> dst (C x R) bf16 --------
// LDS-tiled so both global sides stay coalesced. grid(C/32, R/32, slabs), block(32,8).
__global__ void transpose_cvt_kernel(const float* __restrict__ src,
                                     __bf16* __restrict__ dst, int R, int C) {
  __shared__ __bf16 tile[32][33];
  const float* s = src + (size_t)blockIdx.z * R * C;
  __bf16* d = dst + (size_t)blockIdx.z * R * C;
  const int c0 = blockIdx.x * 32, r0 = blockIdx.y * 32;
#pragma unroll
  for (int i = threadIdx.y; i < 32; i += 8)
    tile[i][threadIdx.x] = (__bf16)s[(size_t)(r0 + i) * C + (c0 + threadIdx.x)];
  __syncthreads();
#pragma unroll
  for (int i = threadIdx.y; i < 32; i += 8)
    d[(size_t)(c0 + i) * R + (r0 + threadIdx.x)] = tile[threadIdx.x][i];
}

// ---------------- LayerNorm (safe_ln) -> bf16 ----------------
__global__ void ln_kernel(const float* __restrict__ x, const float* __restrict__ w,
                          const float* __restrict__ bias, __bf16* __restrict__ out) {
  __shared__ float red[256];
  const int row = blockIdx.x;
  const float* xr = x + (size_t)row * E_;
  float s = 0.f;
  for (int i = threadIdx.x; i < E_; i += 256) s += xr[i];
  red[threadIdx.x] = s; __syncthreads();
  for (int o = 128; o > 0; o >>= 1) {
    if (threadIdx.x < o) red[threadIdx.x] += red[threadIdx.x + o];
    __syncthreads();
  }
  const float mu = red[0] * (1.0f / E_);
  __syncthreads();
  float v = 0.f;
  for (int i = threadIdx.x; i < E_; i += 256) { float d = xr[i] - mu; v += d * d; }
  red[threadIdx.x] = v; __syncthreads();
  for (int o = 128; o > 0; o >>= 1) {
    if (threadIdx.x < o) red[threadIdx.x] += red[threadIdx.x + o];
    __syncthreads();
  }
  const float rstd = rsqrtf(red[0] * (1.0f / E_) + 1e-5f);
  for (int i = threadIdx.x; i < E_; i += 256) {
    float y = (xr[i] - mu) * rstd;
    y = fminf(fmaxf(y, -65000.f), 65000.f);
    out[(size_t)row * E_ + i] = (__bf16)(y * w[i] + bias[i]);
  }
}

// -------- GEMM: C[m][n] = sum_k A_bf16[m][k] * Wb[n][k] (+res), Wb bf16 N-major -------
// grid(M/32, N/512), block 256 (8 waves); each wave: 32(M) x 64(N), K-step 32.
__global__ void gemm_awT(const __bf16* __restrict__ A, const __bf16* __restrict__ Wb,
                         const float* __restrict__ res, float* __restrict__ C,
                         int N, int K) {
  const int lane = threadIdx.x & 31;
  const int wave = threadIdx.x >> 5;
  const int m0 = blockIdx.x * 32;
  const int n0 = blockIdx.y * 512 + wave * 64;
  const int nlo = lane & 15, hl = lane >> 4;
  const __bf16* rowA0 = A + (size_t)(m0 + nlo) * K;
  const __bf16* rowA1 = rowA0 + (size_t)16 * K;
  const __bf16* Wn[4];
#pragma unroll
  for (int s = 0; s < 4; ++s) Wn[s] = Wb + (size_t)(n0 + s * 16 + nlo) * K;

  v8f acc[2][4];
#pragma unroll
  for (int i = 0; i < 2; ++i)
#pragma unroll
    for (int s = 0; s < 4; ++s) acc[i][s] = zero8();

  for (int k0 = 0; k0 < K; k0 += 32) {
    if (k0 + 32 < K) {  // prefetch next K-step A rows (global_prefetch_b8)
      __builtin_prefetch(rowA0 + k0 + 32, 0, 1);
      __builtin_prefetch(rowA1 + k0 + 32, 0, 1);
    }
    v16bf a0 = afrag_bf16(rowA0, k0, lane);
    v16bf a1 = afrag_bf16(rowA1, k0, lane);
#pragma unroll
    for (int s = 0; s < 4; ++s) {
      v16bf b = bfrag_rowT_bf16(Wn[s], k0, lane);
      acc[0][s] = wmma_bf16(a0, b, acc[0][s]);
      acc[1][s] = wmma_bf16(a1, b, acc[1][s]);
    }
  }
#pragma unroll
  for (int i = 0; i < 2; ++i)
#pragma unroll
    for (int s = 0; s < 4; ++s)
#pragma unroll
      for (int r = 0; r < 8; ++r) {
        const int m = m0 + i * 16 + r + 8 * hl;
        const int n = n0 + s * 16 + nlo;
        const size_t off = (size_t)m * N + n;
        float v = acc[i][s][r];
        if (res) v += res[off];
        C[off] = v;
      }
}

// ---------------- RoPE + split into q/k/v bf16, layout [B][NH][T][HD] ----------------
__global__ void rope_kernel(const float* __restrict__ qkv, __bf16* __restrict__ qb,
                            __bf16* __restrict__ kb, __bf16* __restrict__ vb) {
  const int d = threadIdx.x;                 // 0..63
  const int idx = blockIdx.x;                // b*T*NH + t*NH + h
  const int h = idx % NH_;
  const int t = (idx / NH_) % T_;
  const int b = idx / (NH_ * T_);
  const float* base = qkv + (size_t)(b * T_ + t) * (3 * E_) + h * (3 * HD_);
  const float q = base[d], k = base[HD_ + d], v = base[2 * HD_ + d];
  const int fi = (d < 32) ? d : d - 32;
  const float ang = (float)t * powf(10000.0f, -(float)fi * (1.0f / 32.0f));
  float sn, cs;
  sincosf(ang, &sn, &cs);
  float qo, ko;
  if (d < 32) {
    const float q2 = base[d + 32], k2 = base[HD_ + d + 32];
    qo = q * cs - q2 * sn;
    ko = k * cs - k2 * sn;
  } else {
    const float q1 = base[d - 32], k1 = base[HD_ + d - 32];
    qo = q1 * sn + q * cs;
    ko = k1 * sn + k * cs;
  }
  const size_t o = (size_t)((b * NH_ + h) * T_ + t) * HD_ + d;
  qb[o] = (__bf16)qo;
  kb[o] = (__bf16)ko;
  vb[o] = (__bf16)v;
}

// ---------------- Flash attention: grid(T/64, B*NH), block 128 (4 waves) ----------------
__global__ void attn_kernel(const __bf16* __restrict__ qb, const __bf16* __restrict__ kb,
                            const __bf16* __restrict__ vb, __bf16* __restrict__ ctx) {
  __shared__ __bf16 lds[4 * 16 * 32];
  const int lane = threadIdx.x & 31;
  const int wave = threadIdx.x >> 5;
  const int bh = blockIdx.y;
  const int q0 = (blockIdx.x * 4 + wave) * 16;
  const int b = bh / NH_, h = bh % NH_;
  const __bf16* Q  = qb + (size_t)bh * T_ * HD_;
  const __bf16* Kt = kb + (size_t)bh * T_ * HD_;
  const __bf16* V  = vb + (size_t)bh * T_ * HD_;
  __bf16* plds = lds + wave * (16 * 32);
  const int nlo = lane & 15, hl = lane >> 4;

  const __bf16* qrow = Q + (size_t)(q0 + nlo) * HD_;
  const v16bf aQ0 = afrag_bf16(qrow, 0, lane);
  const v16bf aQ1 = afrag_bf16(qrow, 32, lane);

  v8f o[4];
  float mrun[8], lrun[8];
#pragma unroll
  for (int s = 0; s < 4; ++s) o[s] = zero8();
#pragma unroll
  for (int r = 0; r < 8; ++r) { mrun[r] = -1e30f; lrun[r] = 0.f; }

  for (int j0 = 0; j0 < q0 + 16; j0 += 32) {
    // ---- S = Q K^T (16x32 as two 16x16 tiles) ----
    v8f s0 = zero8(), s1 = zero8();
    {
      const __bf16* kr0 = Kt + (size_t)(j0 + nlo) * HD_;
      s0 = wmma_bf16(aQ0, bfrag_rowT_bf16(kr0, 0, lane), s0);
      s0 = wmma_bf16(aQ1, bfrag_rowT_bf16(kr0, 32, lane), s0);
      const __bf16* kr1 = Kt + (size_t)(j0 + 16 + nlo) * HD_;
      s1 = wmma_bf16(aQ0, bfrag_rowT_bf16(kr1, 0, lane), s1);
      s1 = wmma_bf16(aQ1, bfrag_rowT_bf16(kr1, 32, lane), s1);
    }
    const bool need_mask = (j0 + 31 > q0);
    float alpha[8];
#pragma unroll
    for (int r = 0; r < 8; ++r) {
      const int qm = q0 + r + 8 * hl;
      float x0 = s0[r] * 0.125f;           // 1/sqrt(64)
      float x1 = s1[r] * 0.125f;
      if (need_mask) {
        if (j0 + nlo > qm) x0 = -1e30f;
        if (j0 + 16 + nlo > qm) x1 = -1e30f;
      }
      // row-max over the 16 lanes holding this row (WMMA C/D layout)
      float mx = fmaxf(x0, x1);
#pragma unroll
      for (int msk = 1; msk <= 8; msk <<= 1) mx = fmaxf(mx, __shfl_xor(mx, msk, 32));
      const float mnew = fmaxf(mrun[r], mx);
      alpha[r] = expf(mrun[r] - mnew);
      const float p0 = expf(x0 - mnew);
      const float p1 = expf(x1 - mnew);
      s0[r] = p0; s1[r] = p1;
      float ls = p0 + p1;
#pragma unroll
      for (int msk = 1; msk <= 8; msk <<= 1) ls += __shfl_xor(ls, msk, 32);
      lrun[r] = lrun[r] * alpha[r] + ls;
      mrun[r] = mnew;
    }
#pragma unroll
    for (int s = 0; s < 4; ++s)
#pragma unroll
      for (int r = 0; r < 8; ++r) o[s][r] *= alpha[r];

    // ---- transpose P (D-layout -> A-layout) through LDS ----
#pragma unroll
    for (int r = 0; r < 8; ++r) {
      const int m = r + 8 * hl;
      plds[m * 32 + nlo]      = (__bf16)s0[r];
      plds[m * 32 + 16 + nlo] = (__bf16)s1[r];
    }
    asm volatile("s_wait_dscnt 0" ::: "memory");
    const v16bf aP = afrag_bf16(plds + (size_t)nlo * 32, 0, lane);

    // ---- O += P(16x32) * V(32x64) ----
#pragma unroll
    for (int s = 0; s < 4; ++s) {
      v16bf bV = bfrag_col_bf16(V + (size_t)j0 * HD_, HD_, s * 16 + nlo, 0, lane);
      o[s] = wmma_bf16(aP, bV, o[s]);
    }
    asm volatile("s_wait_dscnt 0" ::: "memory");  // aP consumed before next-iter stores
  }

#pragma unroll
  for (int s = 0; s < 4; ++s)
#pragma unroll
    for (int r = 0; r < 8; ++r) {
      const int m = r + 8 * hl;
      const int t = q0 + m;
      const float val = o[s][r] / lrun[r];
      ctx[(size_t)(b * T_ + t) * E_ + h * HD_ + s * 16 + nlo] = (__bf16)val;
    }
}

// ---------------- MoE gate: softmax + top-2 -> score[N][8] (-1 if not selected) -------
__global__ void gate_kernel(const __bf16* __restrict__ h2, const float* __restrict__ gW,
                            float* __restrict__ score) {
  __shared__ float logits[NEXP_];
  const int t = blockIdx.x;
  const int lane = threadIdx.x & 31, wave = threadIdx.x >> 5;   // 8 waves = 8 experts
  const __bf16* hr = h2 + (size_t)t * E_;
  const float* wr = gW + (size_t)wave * E_;
  float s = 0.f;
  for (int k = lane; k < E_; k += 32) s += (float)hr[k] * wr[k];
#pragma unroll
  for (int m = 1; m <= 16; m <<= 1) s += __shfl_xor(s, m, 32);
  if (lane == 0) logits[wave] = s;
  __syncthreads();
  if (threadIdx.x == 0) {
    float mx = logits[0];
    for (int e = 1; e < NEXP_; ++e) mx = fmaxf(mx, logits[e]);
    float se = 0.f, p[NEXP_];
    for (int e = 0; e < NEXP_; ++e) { p[e] = expf(logits[e] - mx); se += p[e]; }
    for (int e = 0; e < NEXP_; ++e) p[e] /= se;
    int i0 = 0;
    for (int e = 1; e < NEXP_; ++e) if (p[e] > p[i0]) i0 = e;
    int i1 = -1;
    for (int e = 0; e < NEXP_; ++e)
      if (e != i0 && (i1 < 0 || p[e] > p[i1])) i1 = e;
    for (int e = 0; e < NEXP_; ++e)
      score[t * NEXP_ + e] = (e == i0 || e == i1) ? p[e] : -1.0f;
  }
}

// ------- per-expert capacity selection: top-CAP_ scores (order within cap irrelevant) --
__global__ void select_kernel(const float* __restrict__ score, int* __restrict__ eidx,
                              float* __restrict__ ewgt) {
  __shared__ int cnt;
  __shared__ int slots;
  const int e = blockIdx.x;
  for (int s = threadIdx.x; s < CAP_; s += blockDim.x) {
    eidx[e * CAP_ + s] = 0;
    ewgt[e * CAP_ + s] = 0.f;
  }
  if (threadIdx.x == 0) { cnt = 0; slots = 0; }
  __syncthreads();
  int lc = 0;
  for (int t = threadIdx.x; t < NTOK_; t += blockDim.x)
    if (score[t * NEXP_ + e] > 0.f) lc++;
  atomicAdd(&cnt, lc);
  __syncthreads();
  const int total = cnt;
  float thr_hi = 0.f, thr_lo = -1.f;
  if (total > CAP_) {
    float lo = 0.f, hi = 1.f;
    for (int it = 0; it < 30; ++it) {
      const float mid = 0.5f * (lo + hi);
      __syncthreads();
      if (threadIdx.x == 0) cnt = 0;
      __syncthreads();
      int c = 0;
      for (int t = threadIdx.x; t < NTOK_; t += blockDim.x)
        if (score[t * NEXP_ + e] > mid) c++;
      atomicAdd(&cnt, c);
      __syncthreads();
      if (cnt > CAP_) lo = mid; else hi = mid;
    }
    thr_hi = hi; thr_lo = lo;
  }
  __syncthreads();
  for (int t = threadIdx.x; t < NTOK_; t += blockDim.x) {
    const float sc = score[t * NEXP_ + e];
    if (sc > thr_hi) {
      const int s = atomicAdd(&slots, 1);
      if (s < CAP_) { eidx[e * CAP_ + s] = t; ewgt[e * CAP_ + s] = sc; }
    }
  }
  __syncthreads();
  if (total > CAP_) {
    for (int t = threadIdx.x; t < NTOK_; t += blockDim.x) {
      const float sc = score[t * NEXP_ + e];
      if (sc > thr_lo && sc <= thr_hi) {
        const int s = atomicAdd(&slots, 1);
        if (s < CAP_) { eidx[e * CAP_ + s] = t; ewgt[e * CAP_ + s] = sc; }
      }
    }
  }
}

// ------- expert GEMM 1: H = gelu(gather(X) @ W1[e] + b1[e]) -> bf16 --------------------
// W1T is bf16 [NEXP][FF][E] (N-major). grid(CAP/32, FF/512, NEXP), block 256.
__global__ void moe_gemm1(const __bf16* __restrict__ h2, const __bf16* __restrict__ W1T,
                          const float* __restrict__ b1, const int* __restrict__ eidx,
                          __bf16* __restrict__ H) {
  const int lane = threadIdx.x & 31, wave = threadIdx.x >> 5;
  const int e = blockIdx.z;
  const int m0 = blockIdx.x * 32;
  const int n0 = blockIdx.y * 512 + wave * 64;
  const int nlo = lane & 15, hl = lane >> 4;
  const __bf16* rowA0 = h2 + (size_t)eidx[e * CAP_ + m0 + nlo] * E_;
  const __bf16* rowA1 = h2 + (size_t)eidx[e * CAP_ + m0 + 16 + nlo] * E_;
  const __bf16* Wn[4];
#pragma unroll
  for (int s = 0; s < 4; ++s)
    Wn[s] = W1T + (size_t)e * FF_ * E_ + (size_t)(n0 + s * 16 + nlo) * E_;

  v8f acc[2][4];
#pragma unroll
  for (int i = 0; i < 2; ++i)
#pragma unroll
    for (int s = 0; s < 4; ++s) acc[i][s] = zero8();

  for (int k0 = 0; k0 < E_; k0 += 32) {
    if (k0 + 32 < E_) {
      __builtin_prefetch(rowA0 + k0 + 32, 0, 1);
      __builtin_prefetch(rowA1 + k0 + 32, 0, 1);
    }
    v16bf a0 = afrag_bf16(rowA0, k0, lane);
    v16bf a1 = afrag_bf16(rowA1, k0, lane);
#pragma unroll
    for (int s = 0; s < 4; ++s) {
      v16bf b = bfrag_rowT_bf16(Wn[s], k0, lane);
      acc[0][s] = wmma_bf16(a0, b, acc[0][s]);
      acc[1][s] = wmma_bf16(a1, b, acc[1][s]);
    }
  }
#pragma unroll
  for (int i = 0; i < 2; ++i)
#pragma unroll
    for (int s = 0; s < 4; ++s)
#pragma unroll
      for (int r = 0; r < 8; ++r) {
        const int m = m0 + i * 16 + r + 8 * hl;
        const int n = n0 + s * 16 + nlo;
        const float x = acc[i][s][r] + b1[e * FF_ + n];
        const float g = 0.5f * x * (1.0f + erff(x * 0.70710678f));  // exact gelu
        H[((size_t)e * CAP_ + m) * FF_ + n] = (__bf16)g;
      }
}

// ------- expert GEMM 2: out += wgt * (H @ W2[e] + b2[e]) scattered by token ------------
// W2T is bf16 [NEXP][E][FF] (N-major). grid(CAP/32, E/512, NEXP), block 256.
__global__ void moe_gemm2(const __bf16* __restrict__ H, const __bf16* __restrict__ W2T,
                          const float* __restrict__ b2, const int* __restrict__ eidx,
                          const float* __restrict__ ewgt, float* __restrict__ out) {
  const int lane = threadIdx.x & 31, wave = threadIdx.x >> 5;
  const int e = blockIdx.z;
  const int m0 = blockIdx.x * 32;
  const int n0 = blockIdx.y * 512 + wave * 64;
  const int nlo = lane & 15, hl = lane >> 4;
  const __bf16* rowA0 = H + ((size_t)e * CAP_ + m0 + nlo) * FF_;
  const __bf16* rowA1 = rowA0 + (size_t)16 * FF_;
  const __bf16* Wn[4];
#pragma unroll
  for (int s = 0; s < 4; ++s)
    Wn[s] = W2T + (size_t)e * E_ * FF_ + (size_t)(n0 + s * 16 + nlo) * FF_;

  v8f acc[2][4];
#pragma unroll
  for (int i = 0; i < 2; ++i)
#pragma unroll
    for (int s = 0; s < 4; ++s) acc[i][s] = zero8();

  for (int k0 = 0; k0 < FF_; k0 += 32) {
    if (k0 + 32 < FF_) {
      __builtin_prefetch(rowA0 + k0 + 32, 0, 1);
      __builtin_prefetch(rowA1 + k0 + 32, 0, 1);
    }
    v16bf a0 = afrag_bf16(rowA0, k0, lane);
    v16bf a1 = afrag_bf16(rowA1, k0, lane);
#pragma unroll
    for (int s = 0; s < 4; ++s) {
      v16bf b = bfrag_rowT_bf16(Wn[s], k0, lane);
      acc[0][s] = wmma_bf16(a0, b, acc[0][s]);
      acc[1][s] = wmma_bf16(a1, b, acc[1][s]);
    }
  }
#pragma unroll
  for (int i = 0; i < 2; ++i)
#pragma unroll
    for (int s = 0; s < 4; ++s)
#pragma unroll
      for (int r = 0; r < 8; ++r) {
        const int m = m0 + i * 16 + r + 8 * hl;
        const float wgt = ewgt[e * CAP_ + m];
        if (wgt > 0.f) {
          const int tok = eidx[e * CAP_ + m];
          const int n = n0 + s * 16 + nlo;
          const float v = (acc[i][s][r] + b2[e * E_ + n]) * wgt;
          atomicAdd(&out[(size_t)tok * E_ + n], v);
        }
      }
}

// ---------------- host-side launch ----------------
extern "C" void kernel_launch(void* const* d_in, const int* in_sizes, int n_in,
                              void* d_out, int out_size, void* d_ws, size_t ws_size,
                              hipStream_t stream) {
  (void)in_sizes; (void)n_in; (void)out_size; (void)ws_size;
  const float* x      = (const float*)d_in[0];
  const float* ln1_w  = (const float*)d_in[1];
  const float* ln1_b  = (const float*)d_in[2];
  const float* Wqkv   = (const float*)d_in[3];
  const float* Wproj  = (const float*)d_in[4];
  const float* ln2_w  = (const float*)d_in[5];
  const float* ln2_b  = (const float*)d_in[6];
  const float* gate_W = (const float*)d_in[7];
  const float* W1     = (const float*)d_in[8];
  const float* b1     = (const float*)d_in[9];
  const float* W2     = (const float*)d_in[10];
  const float* b2     = (const float*)d_in[11];
  float* out = (float*)d_out;

  // workspace carve-up (256B aligned)
  char* p = (char*)d_ws;
  size_t off = 0;
  auto carve = [&](size_t bytes) -> void* {
    void* r = p + off;
    off += (bytes + 255) & ~(size_t)255;
    return r;
  };
  __bf16* h1    = (__bf16*)carve((size_t)NTOK_ * E_ * 2);        // LN1 out
  float*  qkv   = (float*) carve((size_t)NTOK_ * 3 * E_ * 4);    // QKV pre-RoPE
  __bf16* qb    = (__bf16*)carve((size_t)NTOK_ * E_ * 2);        // [B][NH][T][HD]
  __bf16* kbuf  = (__bf16*)carve((size_t)NTOK_ * E_ * 2);
  __bf16* vbuf  = (__bf16*)carve((size_t)NTOK_ * E_ * 2);
  __bf16* ctx   = (__bf16*)carve((size_t)NTOK_ * E_ * 2);        // attention output
  __bf16* h2    = (__bf16*)carve((size_t)NTOK_ * E_ * 2);        // LN2 out
  float*  score = (float*) carve((size_t)NTOK_ * NEXP_ * 4);
  int*    eidx  = (int*)   carve((size_t)NEXP_ * CAP_ * 4);
  float*  ewgt  = (float*) carve((size_t)NEXP_ * CAP_ * 4);
  __bf16* Hh    = (__bf16*)carve((size_t)NEXP_ * CAP_ * FF_ * 2);  // expert hidden
  __bf16* WqkvB = (__bf16*)carve((size_t)3 * E_ * E_ * 2);         // bf16 weights
  __bf16* WprojB= (__bf16*)carve((size_t)E_ * E_ * 2);
  __bf16* W1T   = (__bf16*)carve((size_t)NEXP_ * FF_ * E_ * 2);    // [e][n][k]
  __bf16* W2T   = (__bf16*)carve((size_t)NEXP_ * E_ * FF_ * 2);    // [e][n][k]

  // 0) one-time (per launch) weight conversion / transposition to bf16
  cvt_bf16_kernel<<<(3 * E_ * E_ / 4 + 255) / 256, 256, 0, stream>>>(Wqkv, WqkvB,
                                                                     3 * E_ * E_ / 4);
  cvt_bf16_kernel<<<(E_ * E_ / 4 + 255) / 256, 256, 0, stream>>>(Wproj, WprojB,
                                                                 E_ * E_ / 4);
  transpose_cvt_kernel<<<dim3(FF_ / 32, E_ / 32, NEXP_), dim3(32, 8), 0, stream>>>(
      W1, W1T, E_, FF_);   // src ExFF -> dst FFxE per expert
  transpose_cvt_kernel<<<dim3(E_ / 32, FF_ / 32, NEXP_), dim3(32, 8), 0, stream>>>(
      W2, W2T, FF_, E_);   // src FFxE -> dst ExFF per expert

  // 1) LN1
  ln_kernel<<<NTOK_, 256, 0, stream>>>(x, ln1_w, ln1_b, h1);
  // 2) QKV = h1 @ Wqkv^T   (M=4096, N=3072, K=1024)
  gemm_awT<<<dim3(NTOK_ / 32, 3 * E_ / 512), 256, 0, stream>>>(h1, WqkvB, nullptr, qkv,
                                                               3 * E_, E_);
  // 3) RoPE + split to bf16 heads
  rope_kernel<<<B_ * T_ * NH_, HD_, 0, stream>>>(qkv, qb, kbuf, vbuf);
  // 4) causal flash attention
  attn_kernel<<<dim3(T_ / 64, B_ * NH_), 128, 0, stream>>>(qb, kbuf, vbuf, ctx);
  // 5) x2 = x + ctx @ Wproj^T  -> written straight into d_out (fp32)
  gemm_awT<<<dim3(NTOK_ / 32, E_ / 512), 256, 0, stream>>>(ctx, WprojB, x, out, E_, E_);
  // 6) LN2 on x2
  ln_kernel<<<NTOK_, 256, 0, stream>>>(out, ln2_w, ln2_b, h2);
  // 7) gate softmax + top-2
  gate_kernel<<<NTOK_, 256, 0, stream>>>(h2, gate_W, score);
  // 8) per-expert capacity selection
  select_kernel<<<NEXP_, 256, 0, stream>>>(score, eidx, ewgt);
  // 9) expert FFN up-proj + gelu
  moe_gemm1<<<dim3(CAP_ / 32, FF_ / 512, NEXP_), 256, 0, stream>>>(h2, W1T, b1, eidx, Hh);
  // 10) expert FFN down-proj, weighted scatter-add into d_out
  moe_gemm2<<<dim3(CAP_ / 32, E_ / 512, NEXP_), 256, 0, stream>>>(Hh, W2T, b2, eidx, ewgt,
                                                                  out);
}
